// GNN_9620726743150
// MI455X (gfx1250) — compile-verified, matrix operand order checked by reading, not run on hardware
//
#include <hip/hip_runtime.h>
#include <hip/hip_bf16.h>
#include <math.h>

typedef __attribute__((ext_vector_type(16))) __bf16 v16bf;
typedef __attribute__((ext_vector_type(8)))  float  v8f;

#define F_IN 128
#define HID  128
#define NCLS 16

// ---------------------------------------------------------------------------
// WMMA GEMM: one wave computes a 16 x (16*NT) strip of C = A(MxK) @ B(KxN)
// [+ bias]. A row-major MxK, B row-major KxN, C row-major MxN.
// K is a compile-time constant (multiple of 32); M % 16 == 0; N % (16*NT) == 0.
// bf16 A/B operands (converted in-register from f32), f32 accumulation.
//
// A layout (ISA 7.12.2, 16-bit A 16x32): lane = M%16 (+16 lane-group), VGPR v
//   holds k = (v/4)*16 + (lane/16)*8 + (v%4)*2  (+1 in the high half-word).
// B layout mirrors A with N in place of M.
// C/D layout: lane 0-15 -> N=lane%16, VGPR r -> M = r + 8*(lane/16).
//
// The A fragment is built once per K-step and reused by NT independent WMMAs
// (independent accumulators -> back-to-back XDL issue, no RAW hazard).
// ---------------------------------------------------------------------------
template <int K, int NT>
__global__ __launch_bounds__(32)
void wmma_gemm_bf16(const float* __restrict__ A, const float* __restrict__ B,
                    const float* __restrict__ bias, float* __restrict__ C,
                    int Ncol) {
  const int m0   = blockIdx.x * 16;
  const int n0   = blockIdx.y * (16 * NT);
  const int lane = threadIdx.x & 31;
  const int mr   = lane & 15;
  const int half = lane >> 4;

  const float2* __restrict__ A2 =
      reinterpret_cast<const float2*>(A + (size_t)(m0 + mr) * K);

  v8f acc[NT];
#pragma unroll
  for (int t = 0; t < NT; ++t) acc[t] = v8f{};

#pragma unroll
  for (int kt = 0; kt < K / 32; ++kt) {
    // --- A fragment (shared across all NT output tiles) ---
    v16bf av;
#pragma unroll
    for (int v = 0; v < 8; ++v) {
      const int k0 = (kt << 5) + ((v >> 2) << 4) + (half << 3) + ((v & 3) << 1);
      const float2 a = A2[k0 >> 1];            // global_load_b64
      av[2 * v]     = (__bf16)a.x;
      av[2 * v + 1] = (__bf16)a.y;
    }
    // --- NT independent B fragments / WMMAs ---
#pragma unroll
    for (int t = 0; t < NT; ++t) {
      const int nc = n0 + 16 * t + mr;
      v16bf bv;
#pragma unroll
      for (int v = 0; v < 8; ++v) {
        const int k0 = (kt << 5) + ((v >> 2) << 4) + (half << 3) + ((v & 3) << 1);
        bv[2 * v]     = (__bf16)B[(size_t)k0 * Ncol + nc];
        bv[2 * v + 1] = (__bf16)B[(size_t)(k0 + 1) * Ncol + nc];
      }
      acc[t] = __builtin_amdgcn_wmma_f32_16x16x32_bf16(false, av, false, bv,
                                                       (short)0, acc[t],
                                                       false, false);
    }
  }

#pragma unroll
  for (int t = 0; t < NT; ++t) {
    const int nc = n0 + 16 * t + mr;
    const float badd = bias ? bias[nc] : 0.0f;
#pragma unroll
    for (int r = 0; r < 8; ++r) {
      const int row = m0 + r + (half << 3);
      C[(size_t)row * Ncol + nc] = acc[t][r] + badd;
    }
  }
}

// ---------------------------------------------------------------------------
// GCN normalization helpers
// ---------------------------------------------------------------------------
__global__ void fill_ones(float* p, int n) {
  int i = blockIdx.x * blockDim.x + threadIdx.x;
  if (i < n) p[i] = 1.0f;  // self-loop contributes 1 to every degree
}

__global__ void deg_count(float* deg, const int* __restrict__ dst, int E) {
  int e = blockIdx.x * blockDim.x + threadIdx.x;
  if (e < E) atomicAdd(&deg[dst[e]], 1.0f);
}

__global__ void deg_to_dis(float* p, int n) {
  int i = blockIdx.x * blockDim.x + threadIdx.x;
  if (i < n) p[i] = rsqrtf(p[i]);  // deg >= 1 always (self-loop)
}

// agg[i,f] = bias[f] + t[i,f]*dis[i]^2          (accumulate==0)
// agg[i,f] += bias[f] + t[i,f]*dis[i]^2         (accumulate==1)
__global__ void conv_init(float* __restrict__ agg, const float* __restrict__ t,
                          const float* __restrict__ dis,
                          const float* __restrict__ bias, int total, int accumulate) {
  int i = blockIdx.x * blockDim.x + threadIdx.x;
  if (i >= total) return;
  const int row = i >> 7;       // / HID
  const int f   = i & (HID - 1);
  const float d = dis[row];
  const float v = bias[f] + t[i] * d * d;
  if (accumulate) agg[i] += v;
  else            agg[i] = v;
}

// One block per edge, 128 threads = one feature each (coalesced 512B row;
// the N*H arrays are L2-resident so gathers hit L2 and atomics resolve there).
__global__ __launch_bounds__(HID)
void scatter_edges(float* __restrict__ agg, const float* __restrict__ t,
                   const float* __restrict__ dis,
                   const int* __restrict__ src, const int* __restrict__ dst) {
  const int e = blockIdx.x;
  const int f = threadIdx.x;
  const int s = src[e];
  const int d = dst[e];
  const float w = dis[s] * dis[d];
  atomicAdd(&agg[(size_t)d * HID + f], t[(size_t)s * HID + f] * w);
}

__global__ void relu_ip(float* p, int n) {
  int i = blockIdx.x * blockDim.x + threadIdx.x;
  if (i < n) p[i] = fmaxf(p[i], 0.0f);
}

// log_softmax over 16 classes, one row per thread
__global__ void log_softmax16(const float* __restrict__ logits,
                              float* __restrict__ out, int N) {
  int row = blockIdx.x * blockDim.x + threadIdx.x;
  if (row >= N) return;
  float v[NCLS];
  float mx = -INFINITY;
#pragma unroll
  for (int c = 0; c < NCLS; ++c) {
    v[c] = logits[(size_t)row * NCLS + c];
    mx = fmaxf(mx, v[c]);
  }
  float s = 0.0f;
#pragma unroll
  for (int c = 0; c < NCLS; ++c) s += expf(v[c] - mx);
  const float lse = logf(s);
#pragma unroll
  for (int c = 0; c < NCLS; ++c)
    out[(size_t)row * NCLS + c] = v[c] - mx - lse;
}

// result2/result3: per-row dot products with Wdeg2/Wdeg3 (+ scalar bias)
__global__ void deg_heads(const float* __restrict__ h,
                          const float* __restrict__ w2, const float* __restrict__ b2,
                          const float* __restrict__ w3, const float* __restrict__ b3,
                          float* __restrict__ o2, float* __restrict__ o3, int N) {
  int row = blockIdx.x * blockDim.x + threadIdx.x;
  if (row >= N) return;
  float s2 = b2[0], s3 = b3[0];
  const float* hr = h + (size_t)row * HID;
#pragma unroll 8
  for (int f = 0; f < HID; ++f) {
    const float hv = hr[f];
    s2 += hv * w2[f];
    s3 += hv * w3[f];
  }
  o2[row] = s2;
  o3[row] = s3;
}

// ---------------------------------------------------------------------------
extern "C" void kernel_launch(void* const* d_in, const int* in_sizes, int n_in,
                              void* d_out, int out_size, void* d_ws, size_t ws_size,
                              hipStream_t stream) {
  const float* x     = (const float*)d_in[0];
  const int*   ei    = (const int*)  d_in[1];
  const float* W1    = (const float*)d_in[2];
  const float* b1    = (const float*)d_in[3];
  const float* W2    = (const float*)d_in[4];
  const float* b2    = (const float*)d_in[5];
  const float* Wlin1 = (const float*)d_in[6];
  const float* blin1 = (const float*)d_in[7];
  const float* Wdeg2 = (const float*)d_in[8];
  const float* bdeg2 = (const float*)d_in[9];
  const float* Wdeg3 = (const float*)d_in[10];
  const float* bdeg3 = (const float*)d_in[11];
  const float* Wdown = (const float*)d_in[12];
  const float* bdown = (const float*)d_in[13];

  const int N = in_sizes[0] / F_IN;   // 10000
  const int E = in_sizes[1] / 2;      // 640000
  const int* src = ei;
  const int* dst = ei + E;

  float* out  = (float*)d_out;
  float* out1 = out;                      // [N, C]
  float* out2 = out + (size_t)N * NCLS;   // [N]
  float* out3 = out2 + N;                 // [N]

  // Workspace layout (floats): dis | t | agg | hfin   (~15.4 MB total,
  // resident in the 192 MB L2 across the whole pipeline)
  float* ws   = (float*)d_ws;
  float* dis  = ws;                                   // N
  float* t    = ws + 16384;                           // N*HID
  float* agg  = t   + (size_t)N * HID;                // N*HID
  float* hfin = agg + (size_t)N * HID;                // N*HID (original -> h_final)
  float* logits = agg;                                // reuse after conv2

  const int NH = N * HID;
  const dim3 blk256(256), blk128(HID), blk32(32);
  const dim3 gEl((NH + 255) / 256);
  const dim3 gN((N + 255) / 256);
  const dim3 gE((E + 255) / 256);
  const dim3 gGemmH(N / 16, HID / 64);   // 625 x 2 strips (16 x 64 per wave)
  const dim3 gGemmC(N / 16, 1);          // 625 x 1 tiles  (16 x 16 per wave)

  // --- gcn_norm: deg (with self-loops) -> dis = rsqrt(deg) ---
  fill_ones<<<gN, blk256, 0, stream>>>(dis, N);
  deg_count<<<gE, blk256, 0, stream>>>(dis, dst, E);
  deg_to_dis<<<gN, blk256, 0, stream>>>(dis, N);

  // --- residual: hfin = x @ Wdown + bdown ---
  wmma_gemm_bf16<F_IN, 4><<<gGemmH, blk32, 0, stream>>>(x, Wdown, bdown, hfin, HID);

  // --- conv1: t = x @ W1 ; agg = b1 + self-loop + edge scatter ; ReLU ---
  wmma_gemm_bf16<F_IN, 4><<<gGemmH, blk32, 0, stream>>>(x, W1, nullptr, t, HID);
  conv_init<<<gEl, blk256, 0, stream>>>(agg, t, dis, b1, NH, 0);
  scatter_edges<<<dim3(E), blk128, 0, stream>>>(agg, t, dis, src, dst);
  relu_ip<<<gEl, blk256, 0, stream>>>(agg, NH);

  // --- conv2: t = h @ W2 ; hfin(=original) += b2 + self-loop + scatter ---
  wmma_gemm_bf16<HID, 4><<<gGemmH, blk32, 0, stream>>>(agg, W2, nullptr, t, HID);
  conv_init<<<gEl, blk256, 0, stream>>>(hfin, t, dis, b2, NH, 1);
  scatter_edges<<<dim3(E), blk128, 0, stream>>>(hfin, t, dis, src, dst);

  // --- heads ---
  wmma_gemm_bf16<HID, 1><<<gGemmC, blk32, 0, stream>>>(hfin, Wlin1, blin1, logits, NCLS);
  log_softmax16<<<gN, blk256, 0, stream>>>(logits, out1, N);
  deg_heads<<<gN, blk256, 0, stream>>>(hfin, Wdeg2, bdeg2, Wdeg3, bdeg3, out2, out3, N);
}
